// RNN_77283641524618
// MI455X (gfx1250) — matrix operand order repaired and checked by reading, not compiled, not run on hardware
//
#include <hip/hip_runtime.h>

// ---------------------------------------------------------------------------
// Problem constants (match the reference)
// ---------------------------------------------------------------------------
#define L_ 4
#define D_ 1024
#define H_ 2048
#define S_ 256
#define B_ 128
#define EPS_ 1e-5f

// ---------------------------------------------------------------------------
// CDNA5 WMMA types
// ---------------------------------------------------------------------------
typedef __attribute__((ext_vector_type(16))) __bf16 v16bf;
typedef __attribute__((ext_vector_type(8)))  float  v8f;

union ABfrag { v16bf v; uint4 q[2]; };

__device__ __forceinline__ unsigned short f2bf(float f) {
  // round-to-nearest-even fp32 -> bf16
  unsigned int u = __float_as_uint(f);
  unsigned int r = u + 0x7FFFu + ((u >> 16) & 1u);
  return (unsigned short)(r >> 16);
}

// WGP-scope prefetch at compile-time byte offset OFS from p. Scope field 0
// (WGP) is the encoding default; per the ISA prefetch table only WGP scope
// pulls data into all cache levels (SE/DEV/SYS stop at GL2 — useless here
// since weights are already L2-resident). The clang builtin only emits SE/SYS
// scopes, so use inline asm; the immediate offset lets two streams share one
// base register (no per-iteration v_add_nc_u64, avoiding WMMA->VALU hazards).
template<int OFS>
__device__ __forceinline__ void prefetch_wgp(const void* p) {
  asm volatile("global_prefetch_b8 %0, off offset:%1" :: "v"(p), "n"(OFS));
}

// ---------------------------------------------------------------------------
// fp32 -> bf16 bulk conversion (4 elements / thread)
// ---------------------------------------------------------------------------
__global__ void cvt_f32_to_bf16(const float* __restrict__ src,
                                unsigned short* __restrict__ dst, int n4) {
  int i = blockIdx.x * blockDim.x + threadIdx.x;
  if (i >= n4) return;
  float4 v = ((const float4*)src)[i];
  unsigned short tmp[4] = { f2bf(v.x), f2bf(v.y), f2bf(v.z), f2bf(v.w) };
  ((uint2*)dst)[i] = *(const uint2*)tmp;
}

// ---------------------------------------------------------------------------
// Batched GEMM:  Z[l] = A(_l) * W[l]^T + bias[l]
//   A: (aPerLayer ? L x M x K : M x K) bf16 row-major (contraction contiguous)
//   W: L x N x K bf16 row-major (rows of W are columns of the WMMA B-matrix ->
//      direct contiguous per-lane loads matching the 32x16 bf16 B layout)
//   Z: L x M x N fp32
// Each wave computes a 32x32 output tile (4 f32 accumulators, each A/B frag
// reused twice -> 2 b128 loads per WMMA); 8 waves/block cover 32x256;
// grid = (N/256, M/32, L). K-loop steps one 128B cacheline (64 bf16) with one
// prefetch pair per line.
// ---------------------------------------------------------------------------
template<int K>
__global__ void gemm_rowmajor_bt(const unsigned short* __restrict__ A,
                                 const unsigned short* __restrict__ W,
                                 const float* __restrict__ bias,
                                 float* __restrict__ Z,
                                 int M, int N, int aPerLayer)
{
  const int wave = threadIdx.x >> 5;
  const int lane = threadIdx.x & 31;
  const int half = lane >> 4;     // 0: lanes 0-15, 1: lanes 16-31
  const int lm   = lane & 15;
  const int l    = blockIdx.z;
  const int n0   = (blockIdx.x * 8 + wave) * 32;
  const int m0   = blockIdx.y * 32;

  const unsigned short* abase = A + (aPerLayer ? (size_t)l * M * K : (size_t)0);
  const unsigned short* arow0 = abase + (size_t)(m0 + lm) * K;
  const unsigned short* arow1 = arow0 + (size_t)16 * K;
  const unsigned short* wrow0 = W + (size_t)l * N * K + (size_t)(n0 + lm) * K;
  const unsigned short* wrow1 = wrow0 + (size_t)16 * K;

  const int aoff = half * 16;  // A frag: lanes 16-31 hold K=16..31
  const int boff = half * 16;  // B frag: lanes 16-31 hold K=16..31

  v8f acc00 = {};  // (m-tile 0, n-tile 0)
  v8f acc01 = {};  // (m-tile 0, n-tile 1)
  v8f acc10 = {};  // (m-tile 1, n-tile 0)
  v8f acc11 = {};  // (m-tile 1, n-tile 1)

#pragma unroll 1
  for (int k = 0; k < K; k += 64) {
    // one prefetch per 128B line per weight stream; both share one base reg
    prefetch_wgp<512>(wrow0 + k);               // wrow0 stream, 4 lines ahead
    prefetch_wgp<32 * K + 512>(wrow0 + k);      // wrow1 = wrow0 + 32*K bytes
#pragma unroll
    for (int u = 0; u < 2; ++u) {
      const int kk = k + u * 32;
      ABfrag a0, a1, b0, b1;
      a0.q[0] = *(const uint4*)(arow0 + kk + aoff);
      a0.q[1] = *(const uint4*)(arow0 + kk + aoff + 8);
      a1.q[0] = *(const uint4*)(arow1 + kk + aoff);
      a1.q[1] = *(const uint4*)(arow1 + kk + aoff + 8);
      b0.q[0] = *(const uint4*)(wrow0 + kk + boff);
      b0.q[1] = *(const uint4*)(wrow0 + kk + boff + 8);
      b1.q[0] = *(const uint4*)(wrow1 + kk + boff);
      b1.q[1] = *(const uint4*)(wrow1 + kk + boff + 8);
      acc00 = __builtin_amdgcn_wmma_f32_16x16x32_bf16(false, a0.v, false, b0.v,
                                                      (short)0, acc00, false, false);
      acc01 = __builtin_amdgcn_wmma_f32_16x16x32_bf16(false, a0.v, false, b1.v,
                                                      (short)0, acc01, false, false);
      acc10 = __builtin_amdgcn_wmma_f32_16x16x32_bf16(false, a1.v, false, b0.v,
                                                      (short)0, acc10, false, false);
      acc11 = __builtin_amdgcn_wmma_f32_16x16x32_bf16(false, a1.v, false, b1.v,
                                                      (short)0, acc11, false, false);
    }
  }

  const float bias0 = bias[(size_t)l * N + n0 + lm];
  const float bias1 = bias[(size_t)l * N + n0 + 16 + lm];
  float* zbase = Z + (size_t)l * M * N;
#pragma unroll
  for (int r = 0; r < 8; ++r) {
    const int m = r + half * 8;  // C/D layout: VGPR r -> row r (+8 for hi lanes)
    zbase[(size_t)(m0 + m) * N + n0 + lm]           = acc00[r] + bias0;
    zbase[(size_t)(m0 + m) * N + n0 + 16 + lm]      = acc01[r] + bias1;
    zbase[(size_t)(m0 + 16 + m) * N + n0 + lm]      = acc10[r] + bias0;
    zbase[(size_t)(m0 + 16 + m) * N + n0 + 16 + lm] = acc11[r] + bias1;
  }
}

// ---------------------------------------------------------------------------
// Output GEMM with residual:  v = h3 * Who^T + bho + x ; writes out slice,
// next-step x (fp32, in-place safe: each lane RMWs only its own elements)
// and next-step x_bf (fused conversion). 16x32 wave tile keeps 256 waves of
// parallelism for this small latency-bound GEMM.
// ---------------------------------------------------------------------------
template<int K>
__global__ void gemm_out_res(const unsigned short* __restrict__ A,   // M x K
                             const unsigned short* __restrict__ W,   // N x K
                             const float* __restrict__ bias,         // N
                             float* __restrict__ xres,               // M x N (in/out)
                             unsigned short* __restrict__ xbf,       // M x N
                             float* __restrict__ out,                // M x N
                             int M, int N)
{
  const int wave = threadIdx.x >> 5;
  const int lane = threadIdx.x & 31;
  const int half = lane >> 4;
  const int lm   = lane & 15;
  const int n0   = (blockIdx.x * 8 + wave) * 32;
  const int m0   = blockIdx.y * 16;

  const unsigned short* arow  = A + (size_t)(m0 + lm) * K;
  const unsigned short* wrow0 = W + (size_t)(n0 + lm) * K;
  const unsigned short* wrow1 = wrow0 + (size_t)16 * K;

  const int aoff = half * 16;
  const int boff = half * 16;

  v8f acc0 = {};
  v8f acc1 = {};

#pragma unroll 1
  for (int k = 0; k < K; k += 64) {
    prefetch_wgp<512>(wrow0 + k);
    prefetch_wgp<32 * K + 512>(wrow0 + k);
#pragma unroll
    for (int u = 0; u < 2; ++u) {
      const int kk = k + u * 32;
      ABfrag a, b0, b1;
      a.q[0]  = *(const uint4*)(arow  + kk + aoff);
      a.q[1]  = *(const uint4*)(arow  + kk + aoff + 8);
      b0.q[0] = *(const uint4*)(wrow0 + kk + boff);
      b0.q[1] = *(const uint4*)(wrow0 + kk + boff + 8);
      b1.q[0] = *(const uint4*)(wrow1 + kk + boff);
      b1.q[1] = *(const uint4*)(wrow1 + kk + boff + 8);
      acc0 = __builtin_amdgcn_wmma_f32_16x16x32_bf16(false, a.v, false, b0.v,
                                                     (short)0, acc0, false, false);
      acc1 = __builtin_amdgcn_wmma_f32_16x16x32_bf16(false, a.v, false, b1.v,
                                                     (short)0, acc1, false, false);
    }
  }

  const float bias0 = bias[n0 + lm];
  const float bias1 = bias[n0 + 16 + lm];
#pragma unroll
  for (int r = 0; r < 8; ++r) {
    const int m = m0 + r + half * 8;
    const size_t i0 = (size_t)m * N + n0 + lm;
    const size_t i1 = (size_t)m * N + n0 + 16 + lm;
    const float v0 = acc0[r] + bias0 + xres[i0];
    const float v1 = acc1[r] + bias1 + xres[i1];
    out[i0]  = v0;        out[i1]  = v1;
    xres[i0] = v0;        xres[i1] = v1;
    xbf[i0]  = f2bf(v0);  xbf[i1]  = f2bf(v1);
  }
}

// ---------------------------------------------------------------------------
// Dual LayerNorm + tanh combine: one 256-thread block per (l, b) row of H=2048.
// h_bf[l,b,:] = bf16( tanh( LN(z_ih)*g_ih+be_ih + LN(z_hh)*g_hh+be_hh ) )
// ---------------------------------------------------------------------------
__global__ void ln_tanh_combine(const float* __restrict__ zih, const float* __restrict__ zhh,
                                const float* __restrict__ g_ih, const float* __restrict__ be_ih,
                                const float* __restrict__ g_hh, const float* __restrict__ be_hh,
                                unsigned short* __restrict__ h_bf)
{
  const int b = blockIdx.x, l = blockIdx.y, t = threadIdx.x;
  const size_t row = ((size_t)l * B_ + b) * H_;

  float vi[8], vh[8];
  *(float4*)&vi[0] = *(const float4*)(zih + row + t * 8);
  *(float4*)&vi[4] = *(const float4*)(zih + row + t * 8 + 4);
  *(float4*)&vh[0] = *(const float4*)(zhh + row + t * 8);
  *(float4*)&vh[4] = *(const float4*)(zhh + row + t * 8 + 4);

  float si = 0.f, qi = 0.f, sh = 0.f, qh = 0.f;
#pragma unroll
  for (int j = 0; j < 8; ++j) {
    si += vi[j]; qi += vi[j] * vi[j];
    sh += vh[j]; qh += vh[j] * vh[j];
  }

  __shared__ float red[4][256];
  red[0][t] = si; red[1][t] = qi; red[2][t] = sh; red[3][t] = qh;
  __syncthreads();
  for (int off = 128; off > 0; off >>= 1) {
    if (t < off) {
      red[0][t] += red[0][t + off]; red[1][t] += red[1][t + off];
      red[2][t] += red[2][t + off]; red[3][t] += red[3][t + off];
    }
    __syncthreads();
  }

  const float invH = 1.0f / (float)H_;
  const float mi = red[0][0] * invH, mh = red[2][0] * invH;
  const float ri = rsqrtf(fmaxf(red[1][0] * invH - mi * mi, 0.f) + EPS_);
  const float rh = rsqrtf(fmaxf(red[3][0] * invH - mh * mh, 0.f) + EPS_);

  unsigned short o[8];
#pragma unroll
  for (int j = 0; j < 8; ++j) {
    const size_t gi = (size_t)l * H_ + t * 8 + j;
    const float a = (vi[j] - mi) * ri * g_ih[gi] + be_ih[gi];
    const float c = (vh[j] - mh) * rh * g_hh[gi] + be_hh[gi];
    o[j] = f2bf(tanhf(a + c));
  }
  *(uint4*)(h_bf + row + t * 8) = *(const uint4*)o;
}

// ---------------------------------------------------------------------------
// Host driver: one-time bf16 weight conversion, then 256 sequential steps.
// ---------------------------------------------------------------------------
extern "C" void kernel_launch(void* const* d_in, const int* in_sizes, int n_in,
                              void* d_out, int out_size, void* d_ws, size_t ws_size,
                              hipStream_t stream) {
  (void)in_sizes; (void)n_in; (void)out_size; (void)ws_size;

  const float* x     = (const float*)d_in[0];
  const float* Wih   = (const float*)d_in[1];
  const float* bih   = (const float*)d_in[2];
  const float* g_ih  = (const float*)d_in[3];
  const float* be_ih = (const float*)d_in[4];
  const float* Whh   = (const float*)d_in[5];
  const float* bhh   = (const float*)d_in[6];
  const float* g_hh  = (const float*)d_in[7];
  const float* be_hh = (const float*)d_in[8];
  const float* Who   = (const float*)d_in[9];
  const float* bho   = (const float*)d_in[10];
  float* out = (float*)d_out;

  // Workspace carve-out (~63 MB; all sub-buffers 16B-aligned)
  unsigned short* Wih_bf = (unsigned short*)d_ws;                 // L*H*D bf16
  unsigned short* Whh_bf = Wih_bf + (size_t)L_ * H_ * D_;         // L*H*H bf16
  unsigned short* Who_bf = Whh_bf + (size_t)L_ * H_ * H_;         // D*H   bf16
  unsigned short* h_bf   = Who_bf + (size_t)D_ * H_;              // L*B*H bf16
  unsigned short* x_bf   = h_bf   + (size_t)L_ * B_ * H_;         // B*D   bf16
  float* zih  = (float*)(x_bf + (size_t)B_ * D_);                 // L*B*H f32
  float* zhh  = zih + (size_t)L_ * B_ * H_;                       // L*B*H f32
  float* xcur = zhh + (size_t)L_ * B_ * H_;                       // B*D   f32

  // One-time (per call): weights -> bf16 (L2-resident thereafter), init state
  {
    int n4 = L_ * H_ * D_ / 4;
    cvt_f32_to_bf16<<<(n4 + 255) / 256, 256, 0, stream>>>(Wih, Wih_bf, n4);
    n4 = L_ * H_ * H_ / 4;
    cvt_f32_to_bf16<<<(n4 + 255) / 256, 256, 0, stream>>>(Whh, Whh_bf, n4);
    n4 = D_ * H_ / 4;
    cvt_f32_to_bf16<<<(n4 + 255) / 256, 256, 0, stream>>>(Who, Who_bf, n4);
  }
  hipMemsetAsync(h_bf, 0, (size_t)L_ * B_ * H_ * sizeof(unsigned short), stream);
  hipMemcpyAsync(xcur, x, (size_t)B_ * D_ * sizeof(float),
                 hipMemcpyDeviceToDevice, stream);
  {
    int n4 = B_ * D_ / 4;
    cvt_f32_to_bf16<<<(n4 + 255) / 256, 256, 0, stream>>>(xcur, x_bf, n4);
  }

  // Sequential recurrence
  for (int t = 0; t < S_; ++t) {
    // z_ih = x * Wih^T + bih   (A shared across layers)
    gemm_rowmajor_bt<D_><<<dim3(H_ / 256, B_ / 32, L_), 256, 0, stream>>>(
        x_bf, Wih_bf, bih, zih, B_, H_, 0);
    // z_hh = h * Whh^T + bhh   (A per layer; reads previous-step h_bf)
    gemm_rowmajor_bt<H_><<<dim3(H_ / 256, B_ / 32, L_), 256, 0, stream>>>(
        h_bf, Whh_bf, bhh, zhh, B_, H_, 1);
    // h = tanh(LN(z_ih) + LN(z_hh))
    ln_tanh_combine<<<dim3(B_, L_), 256, 0, stream>>>(
        zih, zhh, g_ih, be_ih, g_hh, be_hh, h_bf);
    // out_t = h[3] * Who^T + bho + x ; feeds back as next x (+ bf16 copy)
    gemm_out_res<H_><<<dim3(D_ / 256, B_ / 16), 256, 0, stream>>>(
        h_bf + (size_t)(L_ - 1) * B_ * H_, Who_bf, bho, xcur, x_bf,
        out + (size_t)t * B_ * D_, B_, D_);
  }
}
